// GraphEncoder_60533269070360
// MI455X (gfx1250) — matrix operand order repaired and checked by reading, not compiled, not run on hardware
//
#include <hip/hip_runtime.h>

#define NNODE   64
#define DMODEL  128
#define NHEAD   4
#define DHEAD   32
#define NLAYER  3
#define DFF     512
#define THREADS 256

typedef __attribute__((ext_vector_type(16))) _Float16 v16h;
typedef __attribute__((ext_vector_type(8)))  float    v8f;

// ---------------- WMMA fragment helpers (wave32, 16x16x32 f16 -> f32) -------
// A (16x32 f16): lanes 0-15 hold row M=lane, halves 0..7 -> K=0..7, halves
// 8..15 -> K=16..23; lanes 16-31 same rows, K offset +8. (ISA 7.12.2)
__device__ __forceinline__ v16h load_fragA(const _Float16* A, int lda, int row0, int kbase) {
  const int lane = threadIdx.x & 31;
  const _Float16* p = A + (row0 + (lane & 15)) * lda + kbase + ((lane >> 4) << 3);
  v16h f;
#pragma unroll
  for (int i = 0; i < 16; ++i) {
    const int k = (i < 8) ? i : (i + 8);
    f[i] = p[k];
  }
  return f;
}

// B operand is staged pre-swizzled into WMMA fragment order:
//   element (k, n) lives at tile=(k>>5)*8+(n>>4), lane=((k&8)<<1)|(n&15), half=(k&7)|((k&16)>>1)
// so each lane's 16 halves are contiguous (32B aligned) -> 2x ds_load_b128.
__device__ __forceinline__ int swzB(int k, int n) {
  return (((((k >> 5) << 3) + (n >> 4)) << 9) | ((k & 8) << 5) | ((n & 15) << 4) | (k & 7) | ((k >> 1) & 8));
}

__device__ __forceinline__ v16h load_fragB_sw(const _Float16* Bsw, int kt, int ct) {
  const int lane = threadIdx.x & 31;
  return *(const v16h*)(Bsw + (((((kt << 3) + ct) << 5) + lane) << 4));
}

// C/D (16x16 f32): VGPR r -> M = r (+8 for lanes 16-31), N = lane & 15.
__device__ __forceinline__ void tile_rowcol(int& rb, int& n, int row0, int col0) {
  const int lane = threadIdx.x & 31;
  n  = col0 + (lane & 15);
  rb = row0 + ((lane >> 4) << 3);
}

__device__ __forceinline__ v8f tile_mm(const _Float16* A, int lda, int row0, int akbase,
                                       const _Float16* Bsw, int ct, int K, v8f c) {
  for (int k0 = 0; k0 < K; k0 += 32) {
    v16h a = load_fragA(A, lda, row0, akbase + k0);
    v16h b = load_fragB_sw(Bsw, k0 >> 5, ct);
    c = __builtin_amdgcn_wmma_f32_16x16x32_f16(false, a, false, b, (short)0, c, false, false);
  }
  return c;
}

// ---------------------------------------------------------------------------
struct Params {
  const float *X, *A0, *projW, *projb, *P, *Q, *alphap;
  const float *ln1s, *ln1b, *linW, *atta, *outW;
  const float *fc1W, *fc1b, *fc2W, *fc2b;
  const float *ln2s, *ln2b, *W1, *b1, *W2, *b2;
  float *outZ, *outMean, *outA;
};

#define SMEM_BYTES 272384

// Stage a 128xC weight block (row-major f32 in global, row stride srcld) into
// swizzled f16 B-fragment layout.
__device__ __forceinline__ void stage_weight_sw(_Float16* dst, const float* src,
                                                int srcld, int t) {
  for (int idx = t; idx < DMODEL * DMODEL; idx += THREADS) {
    const int r = idx >> 7, j = idx & 127;
    dst[swzB(r, j)] = (_Float16)src[r * srcld + j];
  }
}

__device__ __forceinline__ void layernorm_f16(const float* Zin, _Float16* outA,
                                              _Float16* outBsw,
                                              const float* s, const float* b,
                                              float* rowm, float* rowr, int t) {
  if (t < NNODE) {
    float m = 0.f, m2 = 0.f;
    for (int d = 0; d < DMODEL; ++d) { float v = Zin[t * DMODEL + d]; m += v; m2 += v * v; }
    m *= (1.f / DMODEL); m2 *= (1.f / DMODEL);
    rowm[t] = m;
    rowr[t] = rsqrtf(m2 - m * m + 1e-5f);
  }
  __syncthreads();
  for (int idx = t; idx < NNODE * DMODEL; idx += THREADS) {
    const int i = idx >> 7, d = idx & 127;
    const _Float16 v = (_Float16)(((Zin[idx] - rowm[i]) * rowr[i]) * s[d] + b[d]);
    outA[idx] = v;
    if (outBsw) outBsw[swzB(i, d)] = v;  // B-operand copy for A@Hn mix
  }
  __syncthreads();
}

__launch_bounds__(THREADS, 1)
__global__ void graph_encoder_kernel(Params p) {
  const int t    = threadIdx.x;
  const int wave = t >> 5;
  const int bt   = blockIdx.x;

  extern __shared__ unsigned char smem[];
  float*    Zs       = (float*)smem;                 // [64][128] residual / U
  float*    Ys       = Zs + 8192;                    // [64][128] gemm f32 out
  float*    XhS      = Ys + 8192;                    // [64][128] GATv2 features
  float*    As       = XhS + 8192;                   // [64][64] blended A
  float*    logb     = As + 4096;                    // [64][64]
  float*    scratchF = logb + 4096;                  // 64KB: attn scores / H1h
  float*    scores   = scratchF;                     // [256][64]
  _Float16* H1h      = (_Float16*)scratchF;          // [64][512]
  _Float16* Ah       = (_Float16*)(scratchF + 16384);// [64][128] f16 A-operand
  _Float16* Wh       = Ah + 8192;                    // [128][128] swizzled weights
  _Float16* Af16     = Wh + 16384;                   // [64][64] A-operand of mix
  _Float16* Bh       = Af16 + 4096;                  // [64][128] swizzled Hn
  float*    rowm     = (float*)(Bh + 8192);
  float*    rowr     = rowm + 64;
  float*    rowg     = rowr + 64;
  float*    rowsum   = rowg + 64;
  float*    att_s    = rowsum + 64;                  // [4][32]
  unsigned long long* rowmask = (unsigned long long*)(att_s + 128);

  const float alpha = *p.alphap;

  // ---- Blended adjacency A = rownorm(A0 * (1 + alpha*softplus(PQ^T)*mask)) --
  for (int idx = t; idx < NNODE * NNODE; idx += THREADS) {
    const int i = idx >> 6, j = idx & 63;
    float s = 0.f;
#pragma unroll
    for (int r = 0; r < 8; ++r) s += p.P[i * 8 + r] * p.Q[j * 8 + r];
    const float sp = (s > 20.f) ? s : log1pf(__expf(s));
    const float a0 = p.A0[idx];
    const float d  = (a0 > 0.f) ? sp : 0.f;
    As[idx]   = a0 * (1.f + alpha * d);
    logb[idx] = logf(a0 + 1e-8f);
  }
  if (t < NNODE) {
    unsigned long long bits = 0ull;
    for (int j = 0; j < NNODE; ++j)
      if (p.A0[t * NNODE + j] > 0.f) bits |= (1ull << j);
    rowmask[t] = bits;
  }
  __syncthreads();
  if (t < NNODE) {
    float s = 0.f;
    for (int j = 0; j < NNODE; ++j) s += As[t * NNODE + j];
    rowsum[t] = s + 1e-8f;
  }
  __syncthreads();
  for (int idx = t; idx < NNODE * NNODE; idx += THREADS) {
    const float v = As[idx] / rowsum[idx >> 6];
    As[idx]   = v;
    Af16[idx] = (_Float16)v;
    if (bt == 0) p.outA[idx] = v;
  }

  // ---- Z = X @ proj_W + proj_b (WMMA) ------------------------------------
  const float* Xb = p.X + (size_t)bt * NNODE * DMODEL;
  for (int idx = t; idx < NNODE * DMODEL; idx += THREADS) Ah[idx] = (_Float16)Xb[idx];
  stage_weight_sw(Wh, p.projW, DMODEL, t);
  __syncthreads();
  for (int q = 0; q < 4; ++q) {
    v8f c = {};
    c = tile_mm(Ah, DMODEL, q * 16, 0, Wh, wave, DMODEL, c);
    int rb, n; tile_rowcol(rb, n, q * 16, wave * 16);
    const float bias = p.projb[n];
#pragma unroll
    for (int r = 0; r < 8; ++r) Zs[(rb + r) * DMODEL + n] = c[r] + bias;
  }
  __syncthreads();

  // ---- Layers -------------------------------------------------------------
  for (int l = 0; l < NLAYER; ++l) {
    __builtin_prefetch(p.outW + l * 16384, 0, 1);
    __builtin_prefetch(p.W1 + l * 65536, 0, 1);

    // LN1 -> Ah (row-major) + Bh (swizzled B-operand)
    layernorm_f16(Zs, Ah, Bh, p.ln1s + l * DMODEL, p.ln1b + l * DMODEL, rowm, rowr, t);

    // Hmix = A @ Hn  -> Ys  (WMMA, K=64, B from swizzled Bh)
    for (int q = 0; q < 4; ++q) {
      v8f c = {};
      c = tile_mm(Af16, NNODE, q * 16, 0, Bh, wave, NNODE, c);
      int rb, n; tile_rowcol(rb, n, q * 16, wave * 16);
#pragma unroll
      for (int r = 0; r < 8; ++r) Ys[(rb + r) * DMODEL + n] = c[r];
    }
    // stage lin_W
    stage_weight_sw(Wh, p.linW + l * 16384, DMODEL, t);
    __syncthreads();

    // Xh = Hn @ lin_W  (WMMA)
    for (int q = 0; q < 4; ++q) {
      v8f c = {};
      c = tile_mm(Ah, DMODEL, q * 16, 0, Wh, wave, DMODEL, c);
      int rb, n; tile_rowcol(rb, n, q * 16, wave * 16);
#pragma unroll
      for (int r = 0; r < 8; ++r) XhS[(rb + r) * DMODEL + n] = c[r];
    }
    __syncthreads();

    // ---- GATv2 attention: thread = (head h, node i) ----------------------
    for (int idx = t; idx < NHEAD * DHEAD; idx += THREADS) att_s[idx] = p.atta[l * 128 + idx];
    __syncthreads();
    {
      const int h = t >> 6, i = t & 63;
      float* sc = scores + (t << 6);
      const float* xi = XhS + i * DMODEL + h * DHEAD;
      const unsigned long long mbits = rowmask[i];
      float mx = -1e30f;
      for (int j = 0; j < NNODE; ++j) {
        float e = -1e30f;
        if ((mbits >> j) & 1ull) {
          const float* xj = XhS + j * DMODEL + h * DHEAD;
          e = 0.f;
#pragma unroll
          for (int d = 0; d < DHEAD; ++d) {
            float v = xi[d] + xj[d];
            v = (v > 0.f) ? v : 0.2f * v;               // leaky_relu(0.2)
            e += v * att_s[h * DHEAD + d];
          }
          e += logb[i * NNODE + j];
          mx = fmaxf(mx, e);
        }
        sc[j] = e;
      }
      float sum = 0.f;
      for (int j = 0; j < NNODE; ++j) {
        float pr = __expf(sc[j] - mx);
        if (!((mbits >> j) & 1ull)) pr = 0.f;
        sc[j] = pr; sum += pr;
      }
      const float inv = 1.f / sum;
      float T[DHEAD];
#pragma unroll
      for (int d = 0; d < DHEAD; ++d) T[d] = 0.f;
      for (int j = 0; j < NNODE; ++j) {
        const float w = sc[j] * inv;
        const float* xj = XhS + j * DMODEL + h * DHEAD;
#pragma unroll
        for (int d = 0; d < DHEAD; ++d) T[d] += w * xj[d];
      }
#pragma unroll
      for (int d = 0; d < DHEAD; ++d) Ah[i * DMODEL + h * DHEAD + d] = (_Float16)T[d];
    }
    __syncthreads();

    // Y = (attn@Xh) @ out_W, fused add of Hmix (C operand = Ys tile)
    stage_weight_sw(Wh, p.outW + l * 16384, DMODEL, t);
    __syncthreads();
    for (int q = 0; q < 4; ++q) {
      int rb, n; tile_rowcol(rb, n, q * 16, wave * 16);
      v8f c;
#pragma unroll
      for (int r = 0; r < 8; ++r) c[r] = Ys[(rb + r) * DMODEL + n];
      c = tile_mm(Ah, DMODEL, q * 16, 0, Wh, wave, DMODEL, c);
#pragma unroll
      for (int r = 0; r < 8; ++r) Ys[(rb + r) * DMODEL + n] = c[r];
    }
    __syncthreads();

    // ---- NodeGate: U = (Z + Hmix + Y) * g --------------------------------
    if (t < NNODE) {
      float s = 0.f;
      for (int d = 0; d < DMODEL; ++d) s += Zs[t * DMODEL + d] + Ys[t * DMODEL + d];
      s *= (1.f / DMODEL);
      float z = p.fc2b[l];
      for (int d = 0; d < DMODEL; ++d) {
        const float hh = s * p.fc1W[l * DMODEL + d] + p.fc1b[l * DMODEL + d];
        const float si = hh / (1.f + __expf(-hh));      // silu
        z += si * p.fc2W[l * DMODEL + d];
      }
      rowg[t] = 1.f / (1.f + __expf(-z));
    }
    __syncthreads();
    for (int idx = t; idx < NNODE * DMODEL; idx += THREADS)
      Zs[idx] = (Zs[idx] + Ys[idx]) * rowg[idx >> 7];   // Zs now holds U
    __syncthreads();

    // ---- MLP: V = silu(LN2(U)@W1+b1)@W2+b2;  Z = U + V -------------------
    layernorm_f16(Zs, Ah, nullptr, p.ln2s + l * DMODEL, p.ln2b + l * DMODEL, rowm, rowr, t);

    // H1 = silu(Ah @ W1 + b1), streamed in 4 column chunks of 128
    for (int cch = 0; cch < 4; ++cch) {
      stage_weight_sw(Wh, p.W1 + l * (DMODEL * DFF) + cch * 128, DFF, t);
      __syncthreads();
      for (int q = 0; q < 4; ++q) {
        v8f c = {};
        c = tile_mm(Ah, DMODEL, q * 16, 0, Wh, wave, DMODEL, c);
        int rb, n; tile_rowcol(rb, n, q * 16, wave * 16);
        const int gcol = cch * 128 + n;
        const float b1v = p.b1[l * DFF + gcol];
#pragma unroll
        for (int r = 0; r < 8; ++r) {
          float v = c[r] + b1v;
          v = v / (1.f + __expf(-v));                   // silu
          H1h[(rb + r) * DFF + gcol] = (_Float16)v;
        }
      }
      __syncthreads();
    }

    // Z = U + H1 @ W2 + b2, streamed over 4 K chunks of 128
    v8f acc[4];
#pragma unroll
    for (int q = 0; q < 4; ++q) acc[q] = (v8f){};
    for (int kc = 0; kc < 4; ++kc) {
      stage_weight_sw(Wh, p.W2 + l * (DFF * DMODEL) + kc * 16384, DMODEL, t);
      __syncthreads();
      for (int q = 0; q < 4; ++q)
        acc[q] = tile_mm(H1h, DFF, q * 16, kc * 128, Wh, wave, DMODEL, acc[q]);
      __syncthreads();
    }
    for (int q = 0; q < 4; ++q) {
      int rb, n; tile_rowcol(rb, n, q * 16, wave * 16);
      const float b2v = p.b2[l * DMODEL + n];
#pragma unroll
      for (int r = 0; r < 8; ++r) Zs[(rb + r) * DMODEL + n] += acc[q][r] + b2v;
    }
    __syncthreads();
  }

  // ---- Outputs ------------------------------------------------------------
  float* oz = p.outZ + (size_t)bt * NNODE * DMODEL;
  for (int idx = t; idx < NNODE * DMODEL; idx += THREADS) oz[idx] = Zs[idx];
  if (t < DMODEL) {
    float s = 0.f;
    for (int i = 0; i < NNODE; ++i) s += Zs[i * DMODEL + t];
    p.outMean[bt * DMODEL + t] = s * (1.f / NNODE);
  }
}

extern "C" void kernel_launch(void* const* d_in, const int* in_sizes, int n_in,
                              void* d_out, int out_size, void* d_ws, size_t ws_size,
                              hipStream_t stream) {
  (void)in_sizes; (void)n_in; (void)d_ws; (void)ws_size; (void)out_size;
  Params p;
  p.X      = (const float*)d_in[0];
  p.A0     = (const float*)d_in[1];
  // d_in[2] = mask (bool) unused: mask == (A0 > 0) by construction
  p.projW  = (const float*)d_in[3];
  p.projb  = (const float*)d_in[4];
  p.P      = (const float*)d_in[5];
  p.Q      = (const float*)d_in[6];
  p.alphap = (const float*)d_in[7];
  p.ln1s   = (const float*)d_in[8];
  p.ln1b   = (const float*)d_in[9];
  p.linW   = (const float*)d_in[10];
  p.atta   = (const float*)d_in[11];
  p.outW   = (const float*)d_in[12];
  p.fc1W   = (const float*)d_in[13];
  p.fc1b   = (const float*)d_in[14];
  p.fc2W   = (const float*)d_in[15];
  p.fc2b   = (const float*)d_in[16];
  p.ln2s   = (const float*)d_in[17];
  p.ln2b   = (const float*)d_in[18];
  p.W1     = (const float*)d_in[19];
  p.b1     = (const float*)d_in[20];
  p.W2     = (const float*)d_in[21];
  p.b2     = (const float*)d_in[22];

  float* out = (float*)d_out;
  p.outZ    = out;                       // [4,32,64,128]
  p.outMean = out + 1048576;             // [4,32,128]
  p.outA    = out + 1048576 + 16384;     // [64,64]

  hipFuncSetAttribute((const void*)graph_encoder_kernel,
                      hipFuncAttributeMaxDynamicSharedMemorySize, SMEM_BYTES);
  graph_encoder_kernel<<<dim3(128), dim3(THREADS), SMEM_BYTES, stream>>>(p);
}
